// DynamicSparseAttention_86406152061391
// MI455X (gfx1250) — matrix operand order repaired and checked by reading, not compiled, not run on hardware
//
#include <hip/hip_runtime.h>

// ---------------- problem constants ----------------
#define T_SEQ   2048
#define HIDDIM  1024
#define NH      16
#define HD      64
#define SBLK    64
#define NBLK    32
#define MAXK    8

typedef __attribute__((ext_vector_type(16))) __bf16 v16bf;
typedef __attribute__((ext_vector_type(8)))  float  v8f;
typedef __attribute__((ext_vector_type(8)))  unsigned int v8u;
typedef unsigned short u16;
typedef unsigned char  u8;
typedef unsigned int   u32;

// ---------------- bf16 helpers ----------------
__device__ __forceinline__ u16 f32_to_bf16(float f) {
  unsigned int x = __float_as_uint(f);
  x += 0x7fffu + ((x >> 16) & 1u);          // round-to-nearest-even
  return (u16)(x >> 16);
}
__device__ __forceinline__ u32 pack_bf16x2(float lo, float hi) {
  return (u32)f32_to_bf16(lo) | ((u32)f32_to_bf16(hi) << 16);
}
__device__ __forceinline__ v8f wmma_bf16f32(v16bf a, v16bf b, v8f c) {
  return __builtin_amdgcn_wmma_f32_16x16x32_bf16(false, a, false, b,
                                                 (short)0, c, false, false);
}

// ---------------- WMMA fragments from pair-packed u32 LDS tiles ----------
// A fragment 16x32 bf16. LDS tile: row-major, u32 pair p of row r holds
// K=2p (lo16) and K=2p+1 (hi16). ld32 = u32 stride per row.
__device__ __forceinline__ v16bf frag_a32(const u32* base, int ld32, int row0,
                                          int k0, int lane) {
  const u32* rp = base + (row0 + (lane & 15)) * ld32 + (k0 >> 1);
  int half4 = ((lane >> 4) & 1) << 2;
  v8u u;
#pragma unroll
  for (int j = 0; j < 8; ++j)
    u[j] = rp[(j & 3) + half4 + ((j >> 2) << 3)];
  return __builtin_bit_cast(v16bf, u);
}
// B fragment 32x16 bf16. LDS tile: column-major pair-packed: u32 p of column
// n holds K=2p (lo16), K=2p+1 (hi16). ld32 = u32 stride per column.
__device__ __forceinline__ v16bf frag_b32(const u32* base, int ld32, int k0,
                                          int col0, int lane) {
  const u32* cp = base + (col0 + (lane & 15)) * ld32 + (k0 >> 1);
  int pb = ((lane >> 4) & 1) << 2;
  v8u u;
#pragma unroll
  for (int g = 0; g < 8; ++g) u[g] = cp[(g & 3) + pb + ((g >> 2) << 3)];
  return __builtin_bit_cast(v16bf, u);
}
// C/D fragment (16x16 f32) -> LDS row-major f32, stride ld.
__device__ __forceinline__ void store_c(float* base, int ld, int row0, int col0,
                                        int lane, v8f c) {
  int half = (lane >> 4) & 1;
  float* p = base + col0 + (lane & 15);
#pragma unroll
  for (int g = 0; g < 8; ++g) p[(row0 + g + 8 * half) * ld] = c[g];
}

// ---------------- elementwise convert f32 -> packed bf16x2 ----------------
__global__ void cvt_kernel(const float* __restrict__ src, u32* __restrict__ dst,
                           long n2) {
  long i = (long)blockIdx.x * blockDim.x + threadIdx.x;
  long st = (long)gridDim.x * blockDim.x;
  for (; i < n2; i += st) {
    float2 v = *(const float2*)(src + 2 * i);
    dst[i] = pack_bf16x2(v.x, v.y);
  }
}
// Ws (1024x48) -> zero-padded bf16 (1024x64)
__global__ void cvt_ws_kernel(const float* __restrict__ src,
                              u16* __restrict__ dst) {
  int i = blockIdx.x * blockDim.x + threadIdx.x;
  if (i >= HIDDIM * 64) return;
  int r = i >> 6, c = i & 63;
  dst[i] = (c < 48) ? f32_to_bf16(src[r * 48 + c]) : (u16)0;
}

// ---------------- generic LDS-staged bf16 WMMA GEMM (128x64 C block) --------
// C[M,N](f32) = act(alpha * A[M,K](bf16) @ B[K,N](bf16) + bias[N])
// Requirements: M%128==0, K%32==0, lda%8==0, ldb%4==0; B readable for 64 cols
// from each col0 (narrow B matrices are zero-padded). N only guards stores.
// act: 0 none, 1 sigmoid, 2 exact-erf GELU. blockIdx.z batches via strides.
// 8 waves; each wave owns a 2x2 grid of 16x16 tiles (fragment reuse: 2 A-frags
// + 2 B-frags -> 4 independent WMMAs per K-step).
__global__ __launch_bounds__(256) void gemm_bf16_kernel(
    const u16* __restrict__ A, const u16* __restrict__ B, float* __restrict__ C,
    int M, int N, int K, int lda, int ldb, int ldc,
    const float* __restrict__ bias, float alpha, int act,
    long sAz, long sBz, long sCz) {
  __shared__ u32 As32[128 * 20];    // 128 rows x 16 pairs (+4 pad)
  __shared__ u32 BsT32[64 * 17];    // 64 cols x 16 pairs (+1 pad)
  A += (long)blockIdx.z * sAz;
  B += (long)blockIdx.z * sBz;
  C += (long)blockIdx.z * sCz;
  int row0 = blockIdx.y * 128, col0 = blockIdx.x * 64;
  int tid = threadIdx.x, lane = tid & 31, wv = tid >> 5;
  int tm0 = (wv & 3) << 5, tm1 = tm0 + 16;        // row tiles
  int tn0 = (wv >> 2) << 5, tn1 = tn0 + 16;       // col tiles
  v8f acc00, acc01, acc10, acc11;
#pragma unroll
  for (int g = 0; g < 8; ++g) {
    acc00[g] = 0.f; acc01[g] = 0.f; acc10[g] = 0.f; acc11[g] = 0.f;
  }

  // staging assignments
  int ar = tid >> 2, ac8 = (tid & 3) << 3;   // A: 2 x uint4 (rows ar, ar+64)
  int bp = tid >> 4, bc = (tid & 15) << 2;   // B: 4 cols x 2 K rows / thread
  const u16* Aptr0 = A + (size_t)(row0 + ar) * lda + ac8;
  const u16* Aptr1 = A + (size_t)(row0 + ar + 64) * lda + ac8;
  const u16* Bptr0 = B + (size_t)(2 * bp) * ldb + col0 + bc;
  const u16* Bptr1 = Bptr0 + ldb;

  for (int kk = 0; kk < K; kk += 32) {
    uint4 av0 = *(const uint4*)(Aptr0 + kk);
    uint4 av1 = *(const uint4*)(Aptr1 + kk);
    uint2 b0 = *(const uint2*)(Bptr0 + (size_t)kk * ldb);
    uint2 b1 = *(const uint2*)(Bptr1 + (size_t)kk * ldb);
    if (kk + 32 < K) {
      __builtin_prefetch(Aptr0 + kk + 32, 0, 1);
      __builtin_prefetch(Aptr1 + kk + 32, 0, 1);
      __builtin_prefetch(Bptr0 + (size_t)(kk + 32) * ldb, 0, 1);
    }
    *(uint4*)&As32[ar * 20 + (ac8 >> 1)] = av0;
    *(uint4*)&As32[(ar + 64) * 20 + (ac8 >> 1)] = av1;
    BsT32[(bc + 0) * 17 + bp] = (b0.x & 0xffffu) | (b1.x << 16);
    BsT32[(bc + 1) * 17 + bp] = (b0.x >> 16) | (b1.x & 0xffff0000u);
    BsT32[(bc + 2) * 17 + bp] = (b0.y & 0xffffu) | (b1.y << 16);
    BsT32[(bc + 3) * 17 + bp] = (b0.y >> 16) | (b1.y & 0xffff0000u);
    __syncthreads();
    v16bf a0 = frag_a32(As32, 20, tm0, 0, lane);
    v16bf a1 = frag_a32(As32, 20, tm1, 0, lane);
    v16bf bf0 = frag_b32(BsT32, 17, 0, tn0, lane);
    v16bf bf1 = frag_b32(BsT32, 17, 0, tn1, lane);
    acc00 = wmma_bf16f32(a0, bf0, acc00);
    acc01 = wmma_bf16f32(a0, bf1, acc01);
    acc10 = wmma_bf16f32(a1, bf0, acc10);
    acc11 = wmma_bf16f32(a1, bf1, acc11);
    __syncthreads();
  }
  // epilogue (M is always a multiple of 128; only guard columns)
  int half = (lane >> 4) & 1;
  auto epi = [&](v8f a, int tm, int tn) {
    int col = col0 + tn + (lane & 15);
    if (col >= N) return;
    float bv = bias ? bias[col] : 0.f;
#pragma unroll
    for (int g = 0; g < 8; ++g) {
      int row = row0 + tm + g + 8 * half;
      float v = alpha * a[g] + bv;
      if (act == 1)      v = 1.f / (1.f + __expf(-v));
      else if (act == 2) v = 0.5f * v * (1.f + erff(v * 0.70710678118f));
      C[(size_t)row * ldc + col] = v;
    }
  };
  epi(acc00, tm0, tn0);
  epi(acc01, tm0, tn1);
  epi(acc10, tm1, tn0);
  epi(acc11, tm1, tn1);
}

// ---------------- pack compression inputs: (H*NB) x (S*D) ----------------
__global__ void pack_comp_kernel(const u16* __restrict__ k16,
                                 const u16* __restrict__ v16,
                                 u16* __restrict__ Ak, u16* __restrict__ Av) {
  long idx = (long)blockIdx.x * blockDim.x + threadIdx.x;
  if (idx >= 512L * 4096) return;
  int m = (int)(idx >> 12), kk = (int)(idx & 4095);
  int h = m >> 5, nb = m & 31, s = kk >> 6, d = kk & 63;
  size_t src = (size_t)(nb * SBLK + s) * HIDDIM + h * HD + d;
  Ak[idx] = k16[src];
  Av[idx] = v16[src];
}

// -------- kc (H,NB,D f32) -> kcT padded bf16 (H, 64 d-rows, 64 j-cols) -------
__global__ void kct_kernel(const float* __restrict__ kc, u16* __restrict__ kcT) {
  int idx = blockIdx.x * blockDim.x + threadIdx.x;
  if (idx >= NH * 64 * 64) return;
  int h = idx >> 12, d = (idx >> 6) & 63, j = idx & 63;
  kcT[idx] = (j < NBLK) ? f32_to_bf16(kc[((h * NBLK + j) << 6) + d]) : (u16)0;
}

// ---------------- compressed attention: softmax(T x NB) @ vc ----------------
__global__ void comp_out_kernel(const float* __restrict__ imp,
                                const float* __restrict__ vc,
                                float* __restrict__ out) {
  int idx = blockIdx.x * blockDim.x + threadIdx.x;
  if (idx >= NH * T_SEQ) return;
  int h = idx >> 11, t = idx & (T_SEQ - 1);
  const float* row = imp + ((size_t)h * T_SEQ + t) * NBLK;
  float mx = -3.0e38f;
  for (int j = 0; j < NBLK; ++j) mx = fmaxf(mx, row[j]);
  float p[NBLK], sum = 0.f;
  for (int j = 0; j < NBLK; ++j) { p[j] = __expf(row[j] - mx); sum += p[j]; }
  float inv = 1.f / sum;
  for (int d = 0; d < HD; ++d) {
    float s = 0.f;
    for (int j = 0; j < NBLK; ++j) s += p[j] * vc[((h * NBLK + j) << 6) + d];
    out[(size_t)t * HIDDIM + h * HD + d] = s * inv;
  }
}

// ---------------- per-block mean of importance -> block_scores ----------------
__global__ void block_scores_kernel(const float* __restrict__ imp,
                                    float* __restrict__ bsc) {
  int idx = blockIdx.x * blockDim.x + threadIdx.x;
  if (idx >= NH * NBLK * NBLK) return;
  int h = idx >> 10, qb = (idx >> 5) & 31, j = idx & 31;
  float s = 0.f;
  for (int ss = 0; ss < SBLK; ++ss)
    s += imp[(size_t)h * T_SEQ * NBLK + (size_t)(qb * SBLK + ss) * NBLK + j];
  bsc[idx] = s * (1.f / 64.f);
}

// -------- entropy-driven dynamic-k + top-k block mask + aux accumulators ------
__global__ void select_kernel(const float* __restrict__ bsc,
                              const float* __restrict__ kscale,
                              const float* __restrict__ kbias,
                              u8* __restrict__ mask, float* __restrict__ acc) {
  int idx = blockIdx.x * blockDim.x + threadIdx.x;
  if (idx >= NH * NBLK) return;
  int qb = idx & 31;
  const float* row = bsc + (size_t)idx * NBLK;
  float mx = -3.0e38f;
  for (int j = 0; j < NBLK; ++j) mx = fmaxf(mx, row[j]);
  float p[NBLK], sum = 0.f;
  for (int j = 0; j < NBLK; ++j) { p[j] = __expf(row[j] - mx); sum += p[j]; }
  float ent = 0.f;
  for (int j = 0; j < NBLK; ++j) {
    float pr = p[j] / sum;
    ent -= pr * __logf(pr + 1e-9f);
  }
  float norm = ent / (logf((float)NBLK) + 1e-9f);
  float kr = 1.f / (1.f + __expf(-(kscale[0] * norm + kbias[0])));
  float dkf = fminf(fmaxf((float)MAXK * kr, 1.f), (float)MAXK);
  int dk = (int)dkf;  // trunc like torch .long()
  float sc[NBLK];
  for (int j = 0; j < NBLK; ++j) sc[j] = row[j];
  u8 m[NBLK];
  for (int j = 0; j < NBLK; ++j) m[j] = 0;
  for (int kk = 0; kk < MAXK; ++kk) {
    int am = 0; float best = -3.0e38f;
    for (int j = 0; j < NBLK; ++j)
      if (sc[j] > best) { best = sc[j]; am = j; }
    if (kk < dk) m[am] = 1;
    sc[am] = -3.0e38f;
  }
  m[qb] = 1;  // block-diagonal always selected
  for (int j = 0; j < NBLK; ++j) mask[(size_t)idx * NBLK + j] = m[j];
  atomicAdd(&acc[0], (float)dk);
  atomicAdd(&acc[1], (kr - norm) * (kr - norm));
  atomicAdd(&acc[2], kr);
}

// ---------------- flash-style WMMA attention over 64x64 blocks ----------------
// mode 1: sliding window (key blocks {qb-1,qb}, causal diag); mode 0: fine
// block-sparse (key blocks from mask row, no intra-block causality).
// Each wave owns tile row tm and column strip {tn0, tn0+16}: A-fragment reuse.
__global__ __launch_bounds__(256) void attn_kernel(
    int mode, const u16* __restrict__ Qb, const u16* __restrict__ Kb,
    const u16* __restrict__ Vb, const u8* __restrict__ mask,
    float* __restrict__ Out) {
  __shared__ u32 Qs32[64 * 36];     // Q rows, pairs along d
  __shared__ u32 Ks32[64 * 36];     // K rows (key), pairs along d
  __shared__ u32 VsT32[64 * 36];    // V cols (d), pairs along key
  __shared__ u32 Ps32[64 * 36];     // P rows, packed bf16 pairs along key
  __shared__ float Ss[64 * 65];
  __shared__ float mrow[64], lrow[64], scrow[64];
  int qb = blockIdx.x, h = blockIdx.y;
  int tid = threadIdx.x, lane = tid & 31, wv = tid >> 5;
  int tm = (wv & 3) << 4;
  int tn0 = (wv >> 2) << 5, tn1 = tn0 + 16;
  int half = (lane >> 4) & 1;
  v8f o0, o1;
#pragma unroll
  for (int g = 0; g < 8; ++g) { o0[g] = 0.f; o1[g] = 0.f; }

  // stage Q (64x64 bf16): one uint4 (8 elems) per thread, x2
#pragma unroll
  for (int i = 0; i < 2; ++i) {
    int idx = tid + (i << 8);
    int r = idx >> 3, c8 = (idx & 7) << 3;
    uint4 qv = *(const uint4*)(Qb + (size_t)(qb * SBLK + r) * HIDDIM +
                               h * HD + c8);
    *(uint4*)&Qs32[r * 36 + (c8 >> 1)] = qv;
  }
  if (tid < 64) { mrow[tid] = -3.0e38f; lrow[tid] = 0.f; }
  __syncthreads();

  int jstart = (mode == 1) ? (qb > 0 ? qb - 1 : 0) : 0;
  int jend   = (mode == 1) ? qb : (NBLK - 1);
  const u8* mr = mask + ((size_t)h * NBLK + qb) * NBLK;

  for (int j = jstart; j <= jend; ++j) {
    if (mode == 0 && mr[j] == 0) continue;           // uniform per block
    // stage K rows (pairs along d)
#pragma unroll
    for (int i = 0; i < 2; ++i) {
      int idx = tid + (i << 8);
      int r = idx >> 3, c8 = (idx & 7) << 3;
      uint4 kv = *(const uint4*)(Kb + (size_t)(j * SBLK + r) * HIDDIM +
                                 h * HD + c8);
      *(uint4*)&Ks32[r * 36 + (c8 >> 1)] = kv;
    }
    // stage V transposed (cols d, pairs along key)
#pragma unroll
    for (int i = 0; i < 2; ++i) {
      int idx = tid + (i << 8);
      int p = idx >> 4, dg = (idx & 15) << 2;
      const u16* v0 = Vb + (size_t)(j * SBLK + 2 * p) * HIDDIM + h * HD + dg;
      uint2 a = *(const uint2*)v0;
      uint2 b = *(const uint2*)(v0 + HIDDIM);
      VsT32[(dg + 0) * 36 + p] = (a.x & 0xffffu) | (b.x << 16);
      VsT32[(dg + 1) * 36 + p] = (a.x >> 16) | (b.x & 0xffff0000u);
      VsT32[(dg + 2) * 36 + p] = (a.y & 0xffffu) | (b.y << 16);
      VsT32[(dg + 3) * 36 + p] = (a.y >> 16) | (b.y & 0xffff0000u);
    }
    __syncthreads();
    // S = (Q @ K^T) * 1/sqrt(D) : shared A-fragment, two column tiles
    {
      v8f s0, s1;
#pragma unroll
      for (int g = 0; g < 8; ++g) { s0[g] = 0.f; s1[g] = 0.f; }
      for (int k0 = 0; k0 < HD; k0 += 32) {
        v16bf a = frag_a32(Qs32, 36, tm, k0, lane);
        s0 = wmma_bf16f32(a, frag_b32(Ks32, 36, k0, tn0, lane), s0);
        s1 = wmma_bf16f32(a, frag_b32(Ks32, 36, k0, tn1, lane), s1);
      }
#pragma unroll
      for (int g = 0; g < 8; ++g) { s0[g] *= 0.125f; s1[g] *= 0.125f; }
      store_c(Ss, 65, tm, tn0, lane, s0);
      store_c(Ss, 65, tm, tn1, lane, s1);
    }
    __syncthreads();
    // online softmax, one row per thread; P written packed bf16
    bool diag = (mode == 1 && j == qb);
    if (tid < 64) {
      int r = tid;
      float m0 = mrow[r], mx = m0;
      for (int c = 0; c < 64; ++c) {
        float s = Ss[r * 65 + c];
        if (diag && c > r) s = -1.0e30f;
        Ss[r * 65 + c] = s;
        mx = fmaxf(mx, s);
      }
      float scl = __expf(m0 - mx);
      float l = lrow[r] * scl;
      for (int c = 0; c < 64; c += 2) {
        float p0 = __expf(Ss[r * 65 + c] - mx);
        float p1 = __expf(Ss[r * 65 + c + 1] - mx);
        l += p0 + p1;
        Ps32[r * 36 + (c >> 1)] = pack_bf16x2(p0, p1);
      }
      mrow[r] = mx; lrow[r] = l; scrow[r] = scl;
    }
    __syncthreads();
    // O = O*scale + P @ V : shared A-fragment (P), two column tiles
#pragma unroll
    for (int g = 0; g < 8; ++g) {
      float sc = scrow[tm + g + 8 * half];
      o0[g] *= sc;
      o1[g] *= sc;
    }
    for (int k0 = 0; k0 < 64; k0 += 32) {
      v16bf a = frag_a32(Ps32, 36, tm, k0, lane);
      o0 = wmma_bf16f32(a, frag_b32(VsT32, 36, k0, tn0, lane), o0);
      o1 = wmma_bf16f32(a, frag_b32(VsT32, 36, k0, tn1, lane), o1);
    }
    __syncthreads();
  }
  // write Out[t, h*64+d] = O / l  (rows shared by both tiles)
#pragma unroll
  for (int g = 0; g < 8; ++g) {
    int r = tm + g + 8 * half;
    float inv = 1.f / lrow[r];
    size_t rowbase = (size_t)(qb * SBLK + r) * HIDDIM + h * HD;
    Out[rowbase + tn0 + (lane & 15)] = o0[g] * inv;
    Out[rowbase + tn1 + (lane & 15)] = o1[g] * inv;
  }
}

// ---------------- gated combine + packed bf16 for output GEMM ----------------
__global__ void combine_kernel(const float* __restrict__ sl,
                               const float* __restrict__ co,
                               const float* __restrict__ fi,
                               const float* __restrict__ st,
                               u32* __restrict__ merged) {
  long idx = (long)blockIdx.x * blockDim.x + threadIdx.x;
  if (idx >= (long)T_SEQ * HIDDIM / 2) return;
  long e0 = 2 * idx;
  int t = (int)(e0 >> 10), c = (int)(e0 & 1023), h = c >> 6;
  const float* s3 = st + (size_t)t * 48 + h * 3;
  float v0 = sl[e0] * s3[0] + co[e0] * s3[1] + fi[e0] * s3[2];
  float v1 = sl[e0 + 1] * s3[0] + co[e0 + 1] * s3[1] + fi[e0 + 1] * s3[2];
  merged[idx] = pack_bf16x2(v0, v1);
}

__global__ void init_acc_kernel(float* acc) {
  if (threadIdx.x < 3) acc[threadIdx.x] = 0.f;
}
__global__ void finalize_kernel(const float* __restrict__ acc,
                                float* __restrict__ out) {
  if (threadIdx.x == 0) {
    float inv = 1.f / (float)(NH * NBLK);
    out[(size_t)T_SEQ * HIDDIM]     = acc[0] * inv;                    // avg_k
    out[(size_t)T_SEQ * HIDDIM + 1] = 0.01f * (acc[1] * inv) +         // aux
                                      0.01f * (acc[2] * inv);
  }
}

// =================== host-side launch ===================
extern "C" void kernel_launch(void* const* d_in, const int* in_sizes, int n_in,
                              void* d_out, int out_size, void* d_ws,
                              size_t ws_size, hipStream_t stream) {
  (void)in_sizes; (void)n_in; (void)out_size; (void)ws_size;
  const float* x   = (const float*)d_in[0];
  const float* Wq  = (const float*)d_in[1];
  const float* Wk  = (const float*)d_in[2];
  const float* Wv  = (const float*)d_in[3];
  const float* Wo  = (const float*)d_in[4];
  const float* Wc  = (const float*)d_in[5];
  const float* bc  = (const float*)d_in[6];
  const float* ksc = (const float*)d_in[7];
  const float* kbi = (const float*)d_in[8];
  const float* Wsg = (const float*)d_in[9];
  const float* bsg = (const float*)d_in[10];
  float* out = (float*)d_out;

  char* w = (char*)d_ws;
  auto carve = [&](size_t bytes) -> void* {
    void* p = (void*)w;
    w += (bytes + 255) & ~(size_t)255;
    return p;
  };
  const size_t NTOK = (size_t)T_SEQ * HIDDIM;   // 2M
  u16* x16    = (u16*)carve(NTOK * 2);
  u16* Wq16   = (u16*)carve((size_t)HIDDIM * HIDDIM * 2);
  u16* Wk16   = (u16*)carve((size_t)HIDDIM * HIDDIM * 2);
  u16* Wv16   = (u16*)carve((size_t)HIDDIM * HIDDIM * 2);
  u16* Wo16   = (u16*)carve((size_t)HIDDIM * HIDDIM * 2);
  u16* Wc16   = (u16*)carve((size_t)4096 * 64 * 2);
  u16* Ws16p  = (u16*)carve((size_t)HIDDIM * 64 * 2);     // zero-padded to 64
  float* Qp   = (float*)carve(NTOK * 4);
  float* Kp   = (float*)carve(NTOK * 4);
  float* Vp   = (float*)carve(NTOK * 4);
  u16* q16    = (u16*)carve(NTOK * 2);
  u16* k16    = (u16*)carve(NTOK * 2);
  u16* v16    = (u16*)carve(NTOK * 2);
  u16* Ak     = (u16*)carve((size_t)512 * 4096 * 2);
  u16* Av     = (u16*)carve((size_t)512 * 4096 * 2);
  float* kc   = (float*)carve((size_t)512 * 64 * 4);
  float* vc   = (float*)carve((size_t)512 * 64 * 4);
  u16* kcTp   = (u16*)carve((size_t)NH * 64 * 64 * 2);    // zero-padded to 64
  float* imp  = (float*)carve((size_t)NH * T_SEQ * NBLK * 4);
  float* cout = (float*)carve(NTOK * 4);
  float* sout = (float*)carve(NTOK * 4);
  float* fout = (float*)carve(NTOK * 4);
  float* bsc  = (float*)carve((size_t)NH * NBLK * NBLK * 4);
  u8* mask    = (u8*)carve((size_t)NH * NBLK * NBLK);
  float* strat= (float*)carve((size_t)T_SEQ * 48 * 4);
  u16* mrg16  = (u16*)carve(NTOK * 2);
  float* acc  = (float*)carve(256);

  auto cvt = [&](const float* s, u16* d, long n) {
    long n2 = n / 2;
    int blocks = (int)((n2 + 255) / 256);
    if (blocks > 4096) blocks = 4096;
    cvt_kernel<<<dim3(blocks), dim3(256), 0, stream>>>(s, (u32*)d, n2);
  };
  auto gemm = [&](const u16* A, const u16* B, float* C, int M, int N, int K,
                  int lda, int ldb, int ldc, const float* bias, float alpha,
                  int act, int Z, long sA, long sB, long sC) {
    dim3 g((N + 63) / 64, M / 128, Z);
    gemm_bf16_kernel<<<g, dim3(256), 0, stream>>>(A, B, C, M, N, K, lda, ldb,
                                                  ldc, bias, alpha, act, sA,
                                                  sB, sC);
  };

  init_acc_kernel<<<dim3(1), dim3(32), 0, stream>>>(acc);

  // 1) bf16 conversions of inputs / weights
  cvt(x,  x16,  (long)NTOK);
  cvt(Wq, Wq16, (long)HIDDIM * HIDDIM);
  cvt(Wk, Wk16, (long)HIDDIM * HIDDIM);
  cvt(Wv, Wv16, (long)HIDDIM * HIDDIM);
  cvt(Wo, Wo16, (long)HIDDIM * HIDDIM);
  cvt(Wc, Wc16, (long)4096 * 64);
  cvt_ws_kernel<<<dim3((HIDDIM * 64 + 255) / 256), dim3(256), 0, stream>>>(
      Wsg, Ws16p);

  // 2) q/k/v projections (2048x1024x1024 each, WMMA bf16)
  gemm(x16, Wq16, Qp, T_SEQ, HIDDIM, HIDDIM, HIDDIM, HIDDIM, HIDDIM,
       nullptr, 1.f, 0, 1, 0, 0, 0);
  gemm(x16, Wk16, Kp, T_SEQ, HIDDIM, HIDDIM, HIDDIM, HIDDIM, HIDDIM,
       nullptr, 1.f, 0, 1, 0, 0, 0);
  gemm(x16, Wv16, Vp, T_SEQ, HIDDIM, HIDDIM, HIDDIM, HIDDIM, HIDDIM,
       nullptr, 1.f, 0, 1, 0, 0, 0);
  cvt(Qp, q16, (long)NTOK);
  cvt(Kp, k16, (long)NTOK);
  cvt(Vp, v16, (long)NTOK);

  // 3) block compression: gelu((H*NB,4096) @ Wc + bc)
  pack_comp_kernel<<<dim3((int)((512L * 4096 + 255) / 256)), dim3(256), 0,
                     stream>>>(k16, v16, Ak, Av);
  gemm(Ak, Wc16, kc, 512, 64, 4096, 4096, 64, 64, bc, 1.f, 2, 1, 0, 0, 0);
  gemm(Av, Wc16, vc, 512, 64, 4096, 4096, 64, 64, bc, 1.f, 2, 1, 0, 0, 0);
  kct_kernel<<<dim3((NH * 64 * 64 + 255) / 256), dim3(256), 0, stream>>>(
      kc, kcTp);

  // 4) importance = q @ kc^T * scale, batched per head via blockIdx.z
  gemm(q16, kcTp, imp, T_SEQ, NBLK, HD, HIDDIM, 64, NBLK, nullptr, 0.125f, 0,
       NH, (long)HD, (long)64 * 64, (long)T_SEQ * NBLK);

  // 5) compressed-branch output, block scores, dynamic-k selection mask
  comp_out_kernel<<<dim3((NH * T_SEQ + 255) / 256), dim3(256), 0, stream>>>(
      imp, vc, cout);
  block_scores_kernel<<<dim3((NH * NBLK * NBLK + 255) / 256), dim3(256), 0,
                        stream>>>(imp, bsc);
  select_kernel<<<dim3((NH * NBLK + 255) / 256), dim3(256), 0, stream>>>(
      bsc, ksc, kbi, mask, acc);

  // 6) sliding-window attention (mode 1) and block-sparse fine attention (mode 0)
  attn_kernel<<<dim3(NBLK, NH), dim3(256), 0, stream>>>(1, q16, k16, v16, mask,
                                                        sout);
  attn_kernel<<<dim3(NBLK, NH), dim3(256), 0, stream>>>(0, q16, k16, v16, mask,
                                                        fout);

  // 7) strategy gate: sigmoid(x @ Ws + bs), N=48 (B zero-padded to 64)
  gemm(x16, Ws16p, strat, T_SEQ, 48, HIDDIM, HIDDIM, 64, 48, bsg, 1.f, 1, 1,
       0, 0, 0);

  // 8) gated combine -> packed bf16, then final projection into d_out
  combine_kernel<<<dim3((int)((NTOK / 2 + 255) / 256)), dim3(256), 0, stream>>>(
      sout, cout, fout, strat, (u32*)mrg16);
  gemm(mrg16, Wo16, out, T_SEQ, HIDDIM, HIDDIM, HIDDIM, HIDDIM, HIDDIM,
       nullptr, 1.f, 0, 1, 0, 0, 0);

  // 9) scalar outputs (avg_k, aux_loss)
  finalize_kernel<<<dim3(1), dim3(32), 0, stream>>>(acc, out);
}
